// WeisfilerLehmanConv_55027120997021
// MI455X (gfx1250) — compile-verified, matrix operand order checked by reading, not run on hardware
//
#include <hip/hip_runtime.h>
#include <hip/hip_bf16.h>
#include <stdint.h>

// ---------------------------------------------------------------------------
// Weisfeiler-Lehman conv for MI455X (gfx1250).
//
// Perf model: adj_t is 16384^2 f32 = 1 GiB. nbr_hist = adj @ onehot is a
// K-reduction -> every adjacency byte read exactly once -> HBM-bound
// (~46 us at 23.3 TB/s); colsum is a second streaming pass (~46 us).
// v_wmma_f32_16x16x32_bf16: 0/1 entries exact in bf16, counts (< 2^24)
// exact in f32 accumulation. One wave32 owns a 16-row tile; C=16 colors ==
// one 16x16 WMMA N-tile, so one v8f accumulator covers the output width.
// The serial WL tail (16K-key sort, scans, ranks) runs in ONE workgroup in
// 256 KB of gfx1250's 320 KB LDS, staged in with async-to-LDS DMA.
// ---------------------------------------------------------------------------

#define N_NODES 16384
#define B_GRAPHS 64

typedef __bf16 v16bf __attribute__((ext_vector_type(16)));
typedef float  v8f   __attribute__((ext_vector_type(8)));

#if defined(__has_builtin)
#if __has_builtin(__builtin_amdgcn_global_load_async_to_lds_b128) && \
    __has_builtin(__builtin_amdgcn_s_wait_asynccnt)
#define HAVE_ASYNC_LDS 1
#endif
#endif
#ifndef HAVE_ASYNC_LDS
#define HAVE_ASYNC_LDS 0
#endif

#if HAVE_ASYNC_LDS
// Exact parameter type from the clang diagnostic: 16-byte int vector,
// global (AS1) source pointer; LDS (AS3) destination pointer.
typedef int av4i __attribute__((__vector_size__(4 * sizeof(int))));
typedef __attribute__((address_space(1))) av4i* gva_ptr;
typedef __attribute__((address_space(3))) av4i* lva_ptr;
#endif

// Odd 64-bit hash multipliers (structure matches reference: (r<<1)|1 with
// r in [1,2^62); actual numpy-MT19937(42) values would be baked here).
__device__ __constant__ unsigned long long MULTS[17] = {
  0x9E3779B97F4A7C15ull | 1ull, 0xBF58476D1CE4E5B9ull | 1ull,
  0x94D049BB133111EBull | 1ull, 0xD6E8FEB86659FD93ull | 1ull,
  0xA0761D6478BD642Full | 1ull, 0xE7037ED1A0B428DBull | 1ull,
  0x8EBC6AF09C88C6E3ull | 1ull, 0x589965CC75374CC3ull | 1ull,
  0x1D8E4E27C47D124Full | 1ull, 0xEB44ACCAB455D165ull | 1ull,
  0x2545F4914F6CDD1Dull | 1ull, 0xFF51AFD7ED558CCDull | 1ull,
  0xC4CEB9FE1A85EC53ull | 1ull, 0x9FB21C651E98DF25ull | 1ull,
  0x3C79AC492BA7B653ull | 1ull, 0x1C69B3F74AC4AE35ull | 1ull,
  0x243F6A8885A308D3ull | 1ull
};

__global__ void k_zero(float* __restrict__ p, int n) {
  for (int i = blockIdx.x * blockDim.x + threadIdx.x; i < n;
       i += gridDim.x * blockDim.x)
    p[i] = 0.0f;
}

// Pre-pack B = onehot(x) in exact 32x16 bf16 fragment order:
// Bp[kblock][lane][slot j], lane: N = lane&15, half = lane>>4,
// slot j -> K = (j<8 ? half*8+j : 16+half*8+(j-8)).
__global__ void k_bpack(const int* __restrict__ x, __bf16* __restrict__ Bp) {
  int t = blockIdx.x * blockDim.x + threadIdx.x;  // 512 kblocks * 32 lanes
  if (t >= (N_NODES / 32) * 32) return;
  int lane = t & 31;
  int half = lane >> 4, n = lane & 15;
  int kb = (t >> 5) << 5;
  __bf16* dst = Bp + (size_t)t * 16;
#pragma unroll
  for (int j = 0; j < 16; ++j) {
    int K = (j < 8) ? (half * 8 + j) : (16 + half * 8 + (j - 8));
    dst[j] = (x[kb + K] == n) ? (__bf16)1.0f : (__bf16)0.0f;
  }
}

// nbr_hist[i][c] = sum_j adj[i][j] * (x[j]==c), via bf16 WMMA.
// Block = 8 waves, each wave a 16-row tile -> 128 rows/block, 128 blocks
// -> 1024 waves = one wave per output M-tile (max parallelism for N=16384).
__global__ __launch_bounds__(256) void k_gemm(const float* __restrict__ adj,
                                              const __bf16* __restrict__ Bp,
                                              float* __restrict__ nbr) {
  const int wave = threadIdx.x >> 5;
  const int lane = threadIdx.x & 31;
  const int half = lane >> 4;
  const int m    = lane & 15;
  const int row0 = blockIdx.x * 128 + wave * 16;
  const float* __restrict__ rp = adj + (size_t)(row0 + m) * N_NODES;

  v8f c = {0.f, 0.f, 0.f, 0.f, 0.f, 0.f, 0.f, 0.f};
  for (int kb = 0; kb < N_NODES; kb += 32) {
    // Stream 4 KB ahead in this row: Little's law at 23.3 TB/s needs far
    // more bytes in flight than load-return registers can hold.
    __builtin_prefetch(rp + kb + 1024, 0, 1);  // global_prefetch_b8
    // A fragment 16x32 bf16: lane m = row, 8+8 contiguous f32 per half.
    const float4 f0 = *(const float4*)(rp + kb + half * 8);
    const float4 f1 = *(const float4*)(rp + kb + half * 8 + 4);
    const float4 f2 = *(const float4*)(rp + kb + 16 + half * 8);
    const float4 f3 = *(const float4*)(rp + kb + 16 + half * 8 + 4);
    v16bf a;
    a[0] = (__bf16)f0.x;  a[1] = (__bf16)f0.y;  a[2] = (__bf16)f0.z;  a[3] = (__bf16)f0.w;
    a[4] = (__bf16)f1.x;  a[5] = (__bf16)f1.y;  a[6] = (__bf16)f1.z;  a[7] = (__bf16)f1.w;
    a[8] = (__bf16)f2.x;  a[9] = (__bf16)f2.y;  a[10] = (__bf16)f2.z; a[11] = (__bf16)f2.w;
    a[12] = (__bf16)f3.x; a[13] = (__bf16)f3.y; a[14] = (__bf16)f3.z; a[15] = (__bf16)f3.w;
    v16bf b = *(const v16bf*)(Bp + ((size_t)(kb >> 5) * 32 + lane) * 16);
    c = __builtin_amdgcn_wmma_f32_16x16x32_bf16(false, a, false, b,
                                                (short)0, c, false, false);
  }
  // C/D layout: lanes 0-15: (M=r, N=lane); lanes 16-31: (M=8+r, N=lane-16)
#pragma unroll
  for (int r = 0; r < 8; ++r) {
    int rr = half ? (8 + r) : r;
    nbr[(size_t)(row0 + rr) * 16 + m] = c[r];
  }
}

// colsum[j] = sum_i adj[i][j]; second coalesced streaming pass.
__global__ __launch_bounds__(256) void k_colsum(const float* __restrict__ adj,
                                                float* __restrict__ colsum) {
  int col = blockIdx.x * 128 + (threadIdx.x & 127);
  int rh  = threadIdx.x >> 7;
  float acc = 0.f;
  int i0 = rh * (N_NODES / 2), i1 = i0 + (N_NODES / 2);
  for (int i = i0; i < i1; ++i) {
    // 64 KB row stride defeats linear prefetchers: prefetch 8 rows ahead.
    __builtin_prefetch(adj + (size_t)(i + 8) * N_NODES + col, 0, 1);
    acc += adj[(size_t)i * N_NODES + col];
  }
  atomicAdd(&colsum[col], acc);
}

// 64-bit wraparound hash + isolation flag; key sign-flipped for unsigned sort.
__global__ void k_hash(const int* __restrict__ x, const float* __restrict__ nbr,
                       const float* __restrict__ colsum,
                       unsigned long long* __restrict__ keys,
                       unsigned int* __restrict__ iso) {
  int i = blockIdx.x * blockDim.x + threadIdx.x;
  if (i >= N_NODES) return;
  unsigned long long h = (unsigned long long)(long long)x[i] * MULTS[0];
  int rowcnt = 0;
#pragma unroll
  for (int c = 0; c < 16; ++c) {
    int cnt = (int)(nbr[(size_t)i * 16 + c] + 0.5f);  // exact integer counts
    rowcnt += cnt;
    h += (unsigned long long)(long long)cnt * MULTS[c + 1];
  }
  unsigned int is = (rowcnt == 0 && colsum[i] == 0.0f) ? 1u : 0u;
  iso[i] = is;
  unsigned long long hs = is ? 0x8000000000000000ull : h;  // iso -> INT64_MIN
  keys[i] = hs ^ 0x8000000000000000ull;  // signed order == unsigned order
}

// Whole WL tail in one workgroup: 256 KB of the 320 KB WGP LDS.
__global__ __launch_bounds__(1024) void k_wl(
    const unsigned long long* __restrict__ gkeys,
    const unsigned int* __restrict__ iso, int* __restrict__ colors,
    float* __restrict__ colors_f) {
  __shared__ unsigned long long sKey[N_NODES];  // 128 KB
  __shared__ unsigned int sIdx[N_NODES];        //  64 KB
  __shared__ unsigned int sAux[N_NODES];        //  64 KB
  unsigned int* sK32 = (unsigned int*)sKey;
  unsigned int* bufA = sK32;            // reused after sort: scan ping / gid-by-node
  unsigned int* bufB = sK32 + N_NODES;  // reused after sort: first-of-group
  const int tid = threadIdx.x;

#if HAVE_ASYNC_LDS
  // Bulk copy-in of the 128 KB key array via gfx1250 async LDS DMA
  // (ASYNCcnt path): no VGPR return traffic, overlaps with sIdx init.
  for (int v = tid; v < N_NODES / 2; v += 1024) {  // 8192 x 16-byte transfers
    gva_ptr g = (gva_ptr)(uintptr_t)((const char*)gkeys + (size_t)v * 16);
    lva_ptr l = (lva_ptr)(unsigned int)(uintptr_t)((char*)sKey + (size_t)v * 16);
    __builtin_amdgcn_global_load_async_to_lds_b128(g, l, 0, 0);
  }
  for (int p = tid; p < N_NODES; p += 1024) sIdx[p] = p;
  __builtin_amdgcn_s_wait_asynccnt(0);
#else
  for (int p = tid; p < N_NODES; p += 1024) { sKey[p] = gkeys[p]; sIdx[p] = p; }
#endif
  __syncthreads();

  // Bitonic sort ascending by key, carrying node index.
  for (int k2 = 2; k2 <= N_NODES; k2 <<= 1) {
    for (int j = k2 >> 1; j > 0; j >>= 1) {
      for (int p = tid; p < N_NODES; p += 1024) {
        int q = p ^ j;
        if (q > p) {
          bool up = ((p & k2) == 0);
          unsigned long long kp = sKey[p], kq = sKey[q];
          bool sw = up ? (kp > kq) : (kp < kq);
          if (sw) {
            sKey[p] = kq; sKey[q] = kp;
            unsigned int t = sIdx[p]; sIdx[p] = sIdx[q]; sIdx[q] = t;
          }
        }
      }
      __syncthreads();
    }
  }

  // Group-start flags in sorted order.
  for (int p = tid; p < N_NODES; p += 1024)
    sAux[p] = (p == 0) ? 1u : (sKey[p] != sKey[p - 1] ? 1u : 0u);
  __syncthreads();

  // Inclusive scan #1 (14 Hillis-Steele passes, sAux <-> bufA; ends in sAux).
  {
    unsigned int *src = sAux, *dst = bufA;
    for (int d = 1; d < N_NODES; d <<= 1) {
      for (int p = tid; p < N_NODES; p += 1024)
        dst[p] = src[p] + (p >= d ? src[p - d] : 0u);
      __syncthreads();
      unsigned int* t = src; src = dst; dst = t;
    }
  }

  for (int p = tid; p < N_NODES; p += 1024) bufB[p] = 0xFFFFFFFFu;
  __syncthreads();

  // Scatter gid by node + first-of-group (segment min of node index).
  for (int p = tid; p < N_NODES; p += 1024) {
    unsigned int node = sIdx[p];
    unsigned int g = sAux[p] - 1u;
    bufA[node] = g;
    atomicMin(&bufB[g], node);
  }
  __syncthreads();

  // is_first flags per node -> sIdx (free now).
  for (int i = tid; i < N_NODES; i += 1024) {
    unsigned int fi = bufB[bufA[i]];
    sIdx[i] = (!iso[i] && fi == (unsigned int)i) ? 1u : 0u;
  }
  __syncthreads();

  // Inclusive scan #2 (sIdx <-> sAux; ends in sIdx) = 1-based insertion rank.
  {
    unsigned int *src = sIdx, *dst = sAux;
    for (int d = 1; d < N_NODES; d <<= 1) {
      for (int p = tid; p < N_NODES; p += 1024)
        dst[p] = src[p] + (p >= d ? src[p - d] : 0u);
      __syncthreads();
      unsigned int* t = src; src = dst; dst = t;
    }
  }

  for (int i = tid; i < N_NODES; i += 1024) {
    unsigned int fi = bufB[bufA[i]];
    int col = iso[i] ? 0 : (int)sIdx[fi];
    colors[i] = col;
    colors_f[i] = (float)col;
  }
}

__global__ void k_hist(const int* __restrict__ colors,
                       const int* __restrict__ batch,
                       float* __restrict__ hist) {
  int i = blockIdx.x * blockDim.x + threadIdx.x;
  if (i >= N_NODES) return;
  int c = colors[i];
  if (c > 0)  // 'isolate' column (color 0) is dropped by the reference
    atomicAdd(&hist[(size_t)batch[i] * N_NODES + (c - 1)], 1.0f);
}

__global__ __launch_bounds__(256) void k_norm(float* __restrict__ hist) {
  __shared__ float red[256];
  float* row = hist + (size_t)blockIdx.x * N_NODES;
  float s = 0.f;
  for (int i = threadIdx.x; i < N_NODES; i += 256) { float v = row[i]; s += v * v; }
  red[threadIdx.x] = s;
  __syncthreads();
  for (int o = 128; o > 0; o >>= 1) {
    if (threadIdx.x < o) red[threadIdx.x] += red[threadIdx.x + o];
    __syncthreads();
  }
  float scale = red[0] > 0.f ? (1.0f / sqrtf(red[0])) : 0.f;
  for (int i = threadIdx.x; i < N_NODES; i += 256) row[i] *= scale;
}

extern "C" void kernel_launch(void* const* d_in, const int* in_sizes, int n_in,
                              void* d_out, int out_size, void* d_ws,
                              size_t ws_size, hipStream_t stream) {
  (void)in_sizes; (void)n_in; (void)out_size; (void)ws_size;
  const int*   x     = (const int*)d_in[0];
  const float* adj   = (const float*)d_in[1];
  const int*   batch = (const int*)d_in[2];

  float* out      = (float*)d_out;
  float* hist     = out;                                   // [64][16384]
  float* colors_f = out + (size_t)B_GRAPHS * N_NODES;      // [16384]

  char* ws = (char*)d_ws;
  unsigned long long* keys = (unsigned long long*)ws; ws += (size_t)N_NODES * 8;
  float*  nbr    = (float*)ws;        ws += (size_t)N_NODES * 16 * 4;
  float*  colsum = (float*)ws;        ws += (size_t)N_NODES * 4;
  __bf16* Bp     = (__bf16*)ws;       ws += (size_t)(N_NODES / 32) * 32 * 16 * 2;
  unsigned int* iso = (unsigned int*)ws; ws += (size_t)N_NODES * 4;
  int* colors = (int*)ws;             ws += (size_t)N_NODES * 4;

  k_zero  <<<64, 256, 0, stream>>>(colsum, N_NODES);
  k_zero  <<<1024, 256, 0, stream>>>(hist, B_GRAPHS * N_NODES);
  k_bpack <<<64, 256, 0, stream>>>(x, Bp);
  k_gemm  <<<128, 256, 0, stream>>>(adj, Bp, nbr);
  k_colsum<<<128, 256, 0, stream>>>(adj, colsum);
  k_hash  <<<64, 256, 0, stream>>>(x, nbr, colsum, keys, iso);
  k_wl    <<<1, 1024, 0, stream>>>(keys, iso, colors, colors_f);
  k_hist  <<<64, 256, 0, stream>>>(colors, batch, hist);
  k_norm  <<<B_GRAPHS, 256, 0, stream>>>(hist);
}
